// NNBase_71528385348032
// MI455X (gfx1250) — matrix operand order repaired and checked by reading, not compiled, not run on hardware
//
#include <hip/hip_runtime.h>
#include <hip/hip_bf16.h>

// Problem dims (compile-time)
#define T_DIM 128
#define N_DIM 256
#define E_DIM 512
#define A_DIM 8
#define H_DIM 512
#define H3    (3 * H_DIM)
#define EA    (E_DIM + A_DIM)

typedef __attribute__((ext_vector_type(16))) _Float16 v16h;
typedef __attribute__((ext_vector_type(8)))  _Float16 v8h;
typedef __attribute__((ext_vector_type(8)))  float    v8f;

#define WMMA_F16(acc, a, b) \
    (acc) = __builtin_amdgcn_wmma_f32_16x16x32_f16(false, (a), false, (b), (short)0, (acc), false, false)

// ---------------------------------------------------------------------------
// One-time f32 -> f16 staging kernels (parallel, bandwidth bound, off the
// serial recurrence path).
// ---------------------------------------------------------------------------
__global__ void cvt_f32_to_f16_kernel(const float* __restrict__ src,
                                      _Float16* __restrict__ dst, int n) {
    int i = blockIdx.x * blockDim.x + threadIdx.x;
    if (i < n) dst[i] = (_Float16)src[i];
}

// w_ih is (H3 x EA) row-major; keep only the first E columns (A-tail is VALU).
__global__ void cvt_wih_kernel(const float* __restrict__ w_ih,
                               _Float16* __restrict__ dst) {
    int i = blockIdx.x * blockDim.x + threadIdx.x;   // H3*E threads
    int r = i >> 9;           // / E_DIM
    int k = i & (E_DIM - 1);
    dst[i] = (_Float16)w_ih[r * EA + k];
}

// ---------------------------------------------------------------------------
// One GRU timestep for BOTH cells, fused input+recurrent GEMM in WMMA.
//   blocks [0,64)   -> gru   (A = [x_t f16 | h=hx*m_t], all-f16 datapath)
//   blocks [64,128) -> gru_p (A = hy*m_t + g_t*(1-m_t), packed-f16 blend)
// Each wave owns one 16x16 tile of h_new; 3 f32 accumulators (r,z,n gates).
// Stream-order between launches carries the recurrence. New h is written
// both as exact fp32 (d_out) and f16 (ping-pong) for the next step's A side.
// ---------------------------------------------------------------------------
__global__ __launch_bounds__(256)
void gru_step_kernel(int t,
                     const _Float16* __restrict__ xh,      // f16 x (T*N x E)
                     const float* __restrict__ masks,
                     const float* __restrict__ pa,         // prev_action_one_hot
                     const float* __restrict__ gru_init,   // fp32 (epilogue blend)
                     const _Float16* __restrict__ gh16,    // f16 gru_init
                     const float* __restrict__ hxs0,
                     const float* __restrict__ hys0,
                     const _Float16* __restrict__ wih_h,
                     const _Float16* __restrict__ whh_h,
                     const _Float16* __restrict__ whhp_h,
                     const float* __restrict__ w_ih,       // fp32, A-tail cols
                     const float* __restrict__ w_ih_p,
                     const float* __restrict__ b_ih,
                     const float* __restrict__ b_hh,
                     const float* __restrict__ b_ih_p,
                     const float* __restrict__ b_hh_p,
                     _Float16* __restrict__ hx_ping,       // 2 * N*H
                     _Float16* __restrict__ hy_ping,       // 2 * N*H
                     float* __restrict__ outs,
                     float* __restrict__ outs_p) {
    const int wave = threadIdx.x >> 5;
    const int lane = threadIdx.x & 31;
    const int half = lane >> 4;         // 0: lanes 0-15, 1: lanes 16-31
    const int m    = lane & 15;

    const bool is_p = (blockIdx.x >= 64);
    const int  wid  = ((blockIdx.x & 63) << 3) + wave;   // 0..511
    const int  rt   = wid >> 5;                          // 16 row tiles
    const int  ct   = wid & 31;                          // 32 col tiles
    const int  r0   = rt << 4;
    const int  c0   = ct << 4;

    const size_t NH = (size_t)N_DIM * H_DIM;

    // f16 hidden-state source (ping-pong): slot (t-1)&1 == (t+1)&1
    const _Float16* hprev16 = (is_p ? hy_ping : hx_ping) + ((t + 1) & 1) * NH;
    _Float16*       hout16  = (is_p ? hy_ping : hx_ping) + (t & 1) * NH;
    // exact fp32 hidden-state source for the epilogue z*h term
    const float* hprev32 = is_p ? (t ? outs_p + (size_t)(t - 1) * NH : hys0)
                                : (t ? outs   + (size_t)(t - 1) * NH : hxs0);
    float* hout32 = (is_p ? outs_p : outs) + (size_t)t * NH;

    v8f cr = {}, cz = {}, cn = {};

    // ---- A-fragment row carried by this lane (contiguous K, matches B):
    //      elements 0..7  <- K = kk + half*16 + 0..7
    //      elements 8..15 <- K = kk + half*16 + 8 + 0..7
    const int      arow = r0 + m;
    const float    mt_a = masks[t * N_DIM + arow];
    const _Float16 mth  = (_Float16)mt_a;
    const _Float16 omh  = (_Float16)(1.0f - mt_a);

    // ================= phase 1: input GEMM over E (gru only) =================
    if (!is_p) {
        const _Float16* xrow = xh + (size_t)(t * N_DIM + arow) * E_DIM;
        for (int kk = 0; kk < E_DIM; kk += 32) {
            const v8h lo = *(const v8h*)(xrow + kk + half * 16);
            const v8h hi = *(const v8h*)(xrow + kk + half * 16 + 8);
            v16h a;
#pragma unroll
            for (int i = 0; i < 8; ++i) { a[i] = lo[i]; a[8 + i] = hi[i]; }
            {
                v16h b = *(const v16h*)(wih_h + (size_t)(0 * H_DIM + c0 + m) * E_DIM + kk + half * 16);
                WMMA_F16(cr, a, b);
            }
            {
                v16h b = *(const v16h*)(wih_h + (size_t)(1 * H_DIM + c0 + m) * E_DIM + kk + half * 16);
                WMMA_F16(cz, a, b);
            }
            {
                v16h b = *(const v16h*)(wih_h + (size_t)(2 * H_DIM + c0 + m) * E_DIM + kk + half * 16);
                WMMA_F16(cn, a, b);
            }
        }
    }

    // ================= phase 2: recurrent GEMM over H ========================
    {
        const _Float16* whh  = is_p ? whhp_h : whh_h;
        const _Float16* hrow = hprev16 + (size_t)arow * H_DIM;
        const _Float16* grow = gh16 + (size_t)(t * N_DIM + arow) * H_DIM;
        for (int kk = 0; kk < H_DIM; kk += 32) {
            const v8h hlo = *(const v8h*)(hrow + kk + half * 16);
            const v8h hhi = *(const v8h*)(hrow + kk + half * 16 + 8);
            v16h a;
            if (!is_p) {
#pragma unroll
                for (int i = 0; i < 8; ++i) {
                    a[i]     = hlo[i] * mth;          // packed f16 muls
                    a[8 + i] = hhi[i] * mth;
                }
            } else {
                const v8h glo = *(const v8h*)(grow + kk + half * 16);
                const v8h ghi = *(const v8h*)(grow + kk + half * 16 + 8);
#pragma unroll
                for (int i = 0; i < 8; ++i) {
                    a[i]     = hlo[i] * mth + glo[i] * omh;   // packed f16 fma
                    a[8 + i] = hhi[i] * mth + ghi[i] * omh;
                }
            }
            {
                v16h b = *(const v16h*)(whh + (size_t)(0 * H_DIM + c0 + m) * H_DIM + kk + half * 16);
                WMMA_F16(cr, a, b);
            }
            {
                v16h b = *(const v16h*)(whh + (size_t)(1 * H_DIM + c0 + m) * H_DIM + kk + half * 16);
                WMMA_F16(cz, a, b);
            }
            {
                v16h b = *(const v16h*)(whh + (size_t)(2 * H_DIM + c0 + m) * H_DIM + kk + half * 16);
                WMMA_F16(cn, a, b);
            }
        }
    }

    // ================= epilogue: biases, A-tail, gates, store ================
    const int    col = c0 + m;
    const float* bih = is_p ? b_ih_p : b_ih;
    const float* bhh = is_p ? b_hh_p : b_hh;
    const float  bhr = bhh[col], bhz = bhh[H_DIM + col], bhn = bhh[2 * H_DIM + col];
    const float  bir = bih[col], biz = bih[H_DIM + col], bin = bih[2 * H_DIM + col];

#pragma unroll
    for (int g = 0; g < 8; ++g) {
        const int   row = r0 + half * 8 + g;          // C/D layout: M = g + 8*half
        const float mt  = masks[t * N_DIM + row];

        // masked-action contribution (K = 8, pure VALU)
        float gir = bir, giz = biz, gin = bin;
#pragma unroll
        for (int aa = 0; aa < A_DIM; ++aa) {
            const float maA = pa[(size_t)(t * N_DIM + row) * A_DIM + aa] * mt;
            if (!is_p) {
                gir += maA * w_ih[(size_t)(0 * H_DIM + col) * EA + E_DIM + aa];
                giz += maA * w_ih[(size_t)(1 * H_DIM + col) * EA + E_DIM + aa];
                gin += maA * w_ih[(size_t)(2 * H_DIM + col) * EA + E_DIM + aa];
            } else {
                gir += maA * w_ih_p[(0 * H_DIM + col) * A_DIM + aa];
                giz += maA * w_ih_p[(1 * H_DIM + col) * A_DIM + aa];
                gin += maA * w_ih_p[(2 * H_DIM + col) * A_DIM + aa];
            }
        }

        const float ghr = cr[g] + bhr;
        const float ghz = cz[g] + bhz;
        const float ghn = cn[g] + bhn;

        const float r  = 1.0f / (1.0f + __expf(-(gir + ghr)));
        const float z  = 1.0f / (1.0f + __expf(-(giz + ghz)));
        const float s  = gin + r * ghn;
        const float e2 = __expf(-2.0f * s);
        const float nn = (1.0f - e2) / (1.0f + e2);

        const float hp  = hprev32[(size_t)row * H_DIM + col];
        const float hin = is_p
            ? hp * mt + gru_init[(size_t)(t * N_DIM + row) * H_DIM + col] * (1.0f - mt)
            : hp * mt;

        const float hv = (1.0f - z) * nn + z * hin;
        hout32[(size_t)row * H_DIM + col] = hv;
        hout16[(size_t)row * H_DIM + col] = (_Float16)hv;   // next step's A side
    }
}

// ---------------------------------------------------------------------------
// Copy the final hidden states into the hxs / hys output slots.
// ---------------------------------------------------------------------------
__global__ void finalize_kernel(float* __restrict__ out) {
    const size_t TNH = (size_t)T_DIM * N_DIM * H_DIM;
    const size_t NH  = (size_t)N_DIM * H_DIM;
    size_t i = (size_t)blockIdx.x * blockDim.x + threadIdx.x;
    if (i < NH) {
        out[TNH + i]          = out[(size_t)(T_DIM - 1) * NH + i];             // hxs
        out[2 * TNH + NH + i] = out[TNH + NH + (size_t)(T_DIM - 1) * NH + i];  // hys
    }
}

extern "C" void kernel_launch(void* const* d_in, const int* in_sizes, int n_in,
                              void* d_out, int out_size, void* d_ws, size_t ws_size,
                              hipStream_t stream) {
    (void)in_sizes; (void)n_in; (void)out_size; (void)ws_size;

    const float* x        = (const float*)d_in[0];
    const float* hxs0     = (const float*)d_in[1];
    const float* hys0     = (const float*)d_in[2];
    const float* gru_init = (const float*)d_in[3];
    const float* masks    = (const float*)d_in[4];
    const float* pa       = (const float*)d_in[5];
    const float* w_ih     = (const float*)d_in[6];
    const float* w_hh     = (const float*)d_in[7];
    const float* b_ih     = (const float*)d_in[8];
    const float* b_hh     = (const float*)d_in[9];
    const float* w_ih_p   = (const float*)d_in[10];
    const float* w_hh_p   = (const float*)d_in[11];
    const float* b_ih_p   = (const float*)d_in[12];
    const float* b_hh_p   = (const float*)d_in[13];

    float* out = (float*)d_out;
    const size_t TNH = (size_t)T_DIM * N_DIM * H_DIM;
    const size_t NH  = (size_t)N_DIM * H_DIM;
    float* outs   = out;                  // [0, TNH)
    float* outs_p = out + TNH + NH;       // after outs + hxs slot

    // ---- workspace layout (all f16) ----
    _Float16* wih_h   = (_Float16*)d_ws;                          // H3*E
    _Float16* whh_h   = wih_h  + (size_t)H3 * E_DIM;              // H3*H
    _Float16* whhp_h  = whh_h  + (size_t)H3 * H_DIM;              // H3*H
    _Float16* xh      = whhp_h + (size_t)H3 * H_DIM;              // T*N*E
    _Float16* gh16    = xh     + (size_t)T_DIM * N_DIM * E_DIM;   // T*N*H
    _Float16* hx_ping = gh16   + (size_t)T_DIM * N_DIM * H_DIM;   // 2*N*H
    _Float16* hy_ping = hx_ping + 2 * NH;                         // 2*N*H

    // 1) one-time f16 staging (parallel, off the serial path)
    cvt_wih_kernel<<<(H3 * E_DIM) / 256, 256, 0, stream>>>(w_ih, wih_h);
    cvt_f32_to_f16_kernel<<<(H3 * H_DIM) / 256, 256, 0, stream>>>(w_hh, whh_h, H3 * H_DIM);
    cvt_f32_to_f16_kernel<<<(H3 * H_DIM) / 256, 256, 0, stream>>>(w_hh_p, whhp_h, H3 * H_DIM);
    cvt_f32_to_f16_kernel<<<(int)((TNH /*T*N*E==T*N*H*/ + 255) / 256), 256, 0, stream>>>(
        x, xh, (int)((size_t)T_DIM * N_DIM * E_DIM));
    cvt_f32_to_f16_kernel<<<(int)((TNH + 255) / 256), 256, 0, stream>>>(
        gru_init, gh16, (int)TNH);
    // initial hidden states into ping slot 1 (read by t=0: slot (0+1)&1 == 1)
    cvt_f32_to_f16_kernel<<<(int)((NH + 255) / 256), 256, 0, stream>>>(
        hxs0, hx_ping + NH, (int)NH);
    cvt_f32_to_f16_kernel<<<(int)((NH + 255) / 256), 256, 0, stream>>>(
        hys0, hy_ping + NH, (int)NH);

    // 2) 128 sequential fused-GEMM timesteps; stream order carries recurrence
    for (int t = 0; t < T_DIM; ++t) {
        gru_step_kernel<<<128, 256, 0, stream>>>(
            t, xh, masks, pa, gru_init, gh16, hxs0, hys0,
            wih_h, whh_h, whhp_h,
            w_ih, w_ih_p, b_ih, b_hh, b_ih_p, b_hh_p,
            hx_ping, hy_ping, outs, outs_p);
    }

    // 3) copy last hidden states into hxs/hys slots
    finalize_kernel<<<(int)((NH + 255) / 256), 256, 0, stream>>>(out);
}
